// SSMRTF_model_24292335026641
// MI455X (gfx1250) — matrix operand order repaired and checked by reading, not compiled, not run on hardware
//
#include <hip/hip_runtime.h>
#include <stdint.h>

// ---------------------------------------------------------------------------
// RTF (order-64 IIR) long conv, reformulated as a 512-tap truncated FIR
// (exact in fp32: spectral radius ~0.01^(1/64)=0.93, 0.93^512 ~ 4e-17),
// executed as per-channel Toeplitz GEMMs on v_wmma_f32_16x16x32_bf16.
//   d_in[0] = x  (4, 8192, 512) f32
//   d_in[1] = A  (512, 64)      f32
//   d_in[2] = B  (512, 64)      f32
//   d_in[3] = h0 (1,)           f32
//   d_out   = gelu(conv(x,K) + h0*x)  (4, 8192, 512) f32
// Workspace: KR (512*640 bf16 = 640KB) + xT (4*512*8192 bf16 = 32MB)
// ---------------------------------------------------------------------------

#define LSEQ   8192
#define NCH    512
#define NB     4
#define ORDER  64
#define KTAPS  512           // truncated impulse-response length
#define WIN    576           // GEMM K-dim per sub-tile (= KTAPS + 64)
#define KRLEN  640           // reversed, zero-padded kernel row length
#define XWIN   768           // union of the 4 sub-tile windows per (ch, batch)
#define TSUP   256           // output timesteps per workgroup
#define CHG    8             // channels per workgroup (1 per wave)
#define KT_TILES (WIN / 32)  // 18

typedef __attribute__((ext_vector_type(16))) __bf16 v16bf;
typedef __attribute__((ext_vector_type(8)))  float  v8f;

union FragU {
  unsigned short s[16];
  uint4          q[2];
  v16bf          v;
};

__device__ __forceinline__ unsigned short f2bf(float f) {
  union { float f; unsigned int u; } c; c.f = f;
  unsigned int u = c.u + 0x7fffu + ((c.u >> 16) & 1u);   // RNE
  return (unsigned short)(u >> 16);
}
__device__ __forceinline__ float bf2f(unsigned short h) {
  union { float f; unsigned int u; } c; c.u = ((unsigned int)h) << 16;
  return c.f;
}
__device__ __forceinline__ float gelu_tanh(float x) {
  float t = tanhf(0.7978845608028654f * (x + 0.044715f * x * x * x));
  return 0.5f * x * (1.0f + t);
}

// ---------------------------------------------------------------------------
// Kernel 1: per-channel impulse response K[t] = B[t]1{t<64} - sum A[i] K[t-i],
// stored reversed + zero-padded as bf16:  KR[q] = K[574 - q] for q in [63,574].
// Then Toeplitz A[m,w] = KR[w + 63 - m] is a contiguous gather.
// ---------------------------------------------------------------------------
__global__ void __launch_bounds__(256)
rtf_build_kernel(const float* __restrict__ A, const float* __restrict__ Bc,
                 unsigned short* __restrict__ KRg) {
  int c = blockIdx.x * blockDim.x + threadIdx.x;
  if (c >= NCH) return;
  float a[ORDER];
#pragma unroll
  for (int i = 0; i < ORDER; ++i) a[i] = A[c * ORDER + i];
  float hist[ORDER];
#pragma unroll
  for (int i = 0; i < ORDER; ++i) hist[i] = 0.0f;

  unsigned short* kr = KRg + (size_t)c * KRLEN;
  for (int q = 0; q < 63; ++q)       kr[q] = 0;
  for (int q = 575; q < KRLEN; ++q)  kr[q] = 0;

  for (int blk = 0; blk < KTAPS / ORDER; ++blk) {   // 8 blocks of 64 taps
    float cur[ORDER];
#pragma unroll
    for (int t = 0; t < ORDER; ++t) {
      float s = (blk == 0) ? Bc[c * ORDER + t] : 0.0f;
#pragma unroll
      for (int i = 1; i <= ORDER; ++i) {            // static ring indexing
        float prev = (t - i >= 0) ? cur[t - i] : hist[ORDER + t - i];
        s = fmaf(-a[i - 1], prev, s);
      }
      cur[t] = s;
      kr[574 - (blk * ORDER + t)] = f2bf(s);
    }
#pragma unroll
    for (int i = 0; i < ORDER; ++i) hist[i] = cur[i];
  }
}

// ---------------------------------------------------------------------------
// Kernel 2: x (B,L,C) f32  ->  xT (B,C,L) bf16, 64x64 LDS tiles, both sides
// coalesced. Makes conv windows contiguous per channel.
// ---------------------------------------------------------------------------
__global__ void __launch_bounds__(256)
rtf_xpose_kernel(const float* __restrict__ x, unsigned short* __restrict__ xT) {
  __shared__ unsigned short tile[64][65];
  int b  = blockIdx.z;
  int t0 = blockIdx.x * 64;
  int c0 = blockIdx.y * 64;
  int cl = threadIdx.x & 63, tq = threadIdx.x >> 6;
#pragma unroll
  for (int i = 0; i < 16; ++i) {
    int t = tq + i * 4;
    tile[t][cl] = f2bf(x[((size_t)b * LSEQ + (t0 + t)) * NCH + c0 + cl]);
  }
  __syncthreads();
  int tf = threadIdx.x & 63, cq = threadIdx.x >> 6;
#pragma unroll
  for (int i = 0; i < 16; ++i) {
    int cc = cq + i * 4;
    xT[((size_t)b * NCH + (c0 + cc)) * LSEQ + t0 + tf] = tile[tf][cc];
  }
}

// ---------------------------------------------------------------------------
// Kernel 3: WMMA conv + h0*x + GELU.
// Per wave: one channel, GEMM M=64 (t in sub-tile), N=16 (4 batches x 4
// sub-tiles), K=576; A = Toeplitz(K) gathered from LDS, B = x windows in LDS.
// ---------------------------------------------------------------------------
union ConvShm {                         // 48KB; o-phase reuses x-phase storage
  unsigned short x[CHG][NB][XWIN];      // bf16 windows: t in [t0-511, t0+256]
  float          o[TSUP * NB * CHG];    // output staging for coalesced stores
};

__device__ __forceinline__ v16bf
load_a_frag(const unsigned short* __restrict__ krow, int mt, int kt,
            int mrow, int half) {
  // A[m,w] = KR[w + 63 - m], fragment layout per CDNA5 16-bit A (16x32):
  // lanes 0-15: K=0..7,16..23 ; lanes 16-31: K=8..15,24..31 (pairs per VGPR).
  // The 4 mt fragments overlap by 16 halves; the compiler CSEs the loads
  // into a rolling set of ds_load_b128s.
  FragU fa;
  int base = kt * 32 + half * 8 + 63 - (mt * 16 + mrow);
#pragma unroll
  for (int i = 0; i < 16; ++i) {
    int v = i >> 1, pos = i & 1;
    int k = ((v < 4) ? (v << 1) : (16 + ((v - 4) << 1))) + pos;
    fa.s[i] = krow[base + k];
  }
  return fa.v;
}

__global__ void __launch_bounds__(256)
rtf_conv_kernel(const unsigned short* __restrict__ xT,
                const unsigned short* __restrict__ KRg,
                const float* __restrict__ h0p,
                float* __restrict__ out) {
  __shared__ __align__(16) ConvShm sU;
  __shared__ unsigned short sKR[CHG][KRLEN];        // 10KB

  const int tid = threadIdx.x;
  const int t0  = blockIdx.x * TSUP;
  const int c0  = blockIdx.y * CHG;

  // Stage reversed kernels (coalesced-ish)
  for (int e = tid; e < CHG * KRLEN; e += 256) {
    int ci = e / KRLEN, q = e - ci * KRLEN;
    sKR[ci][q] = KRg[(size_t)(c0 + ci) * KRLEN + q];
  }
  // Stage x windows, zero-fill outside [0, L)
  for (int e = tid; e < CHG * NB * XWIN; e += 256) {
    int row = e / XWIN, u = e - row * XWIN;
    int ci = row >> 2, b = row & 3;
    int t = t0 - 511 + u;
    unsigned short v = 0;
    if (t >= 0 && t < LSEQ)
      v = xT[((size_t)b * NCH + (c0 + ci)) * LSEQ + t];
    sU.x[ci][b][u] = v;
  }
  // L2 streaming hint for the next supertile's windows (xT is L2-resident:
  // 32MB << 192MB global L2). One cacheline hint per (ch, batch) row.
  if (tid < CHG * NB) {
    int ci = tid >> 2, b = tid & 3;
    int tn = t0 + TSUP - 511;
    if (tn >= 0 && tn < LSEQ - TSUP)
      __builtin_prefetch(&xT[((size_t)b * NCH + (c0 + ci)) * LSEQ + tn], 0, 1);
  }
  __syncthreads();

  const int lane = tid & 31;
  const int ci   = tid >> 5;           // one channel per wave
  const int mrow = lane & 15;
  const int half = lane >> 4;
  const int ncol = lane & 15;          // N column: 4*subtile + batch
  const int subt = ncol >> 2;
  const int bn   = ncol & 3;
  const unsigned short* krow = &sKR[ci][0];

  v8f acc[4];
#pragma unroll
  for (int mt = 0; mt < 4; ++mt) {
#pragma unroll
    for (int j = 0; j < 8; ++j) acc[mt][j] = 0.0f;
  }

  for (int kt = 0; kt < KT_TILES; ++kt) {
    // B fragment (32x16 bf16): lane holds column ncol, window elements at
    // k = kt*32 + half*8 + {0..7, 16..23}  ->  two 16B LDS loads.
    FragU fb;
    int ubase = subt * 64 + kt * 32 + half * 8;
    const uint4* bp = reinterpret_cast<const uint4*>(&sU.x[ci][bn][ubase]);
    fb.q[0] = bp[0];
    fb.q[1] = bp[2];

    v16bf a0 = load_a_frag(krow, 0, kt, mrow, half);
    v16bf a1 = load_a_frag(krow, 1, kt, mrow, half);
    v16bf a2 = load_a_frag(krow, 2, kt, mrow, half);
    v16bf a3 = load_a_frag(krow, 3, kt, mrow, half);

    acc[0] = __builtin_amdgcn_wmma_f32_16x16x32_bf16(
        false, a0, false, fb.v, (short)0, acc[0], false, false);
    acc[1] = __builtin_amdgcn_wmma_f32_16x16x32_bf16(
        false, a1, false, fb.v, (short)0, acc[1], false, false);
    acc[2] = __builtin_amdgcn_wmma_f32_16x16x32_bf16(
        false, a2, false, fb.v, (short)0, acc[2], false, false);
    acc[3] = __builtin_amdgcn_wmma_f32_16x16x32_bf16(
        false, a3, false, fb.v, (short)0, acc[3], false, false);
  }

  // Epilogue: + h0*x, GELU (read x from LDS window before it is reused)
  const float h0v = h0p[0];
#pragma unroll
  for (int mt = 0; mt < 4; ++mt) {
#pragma unroll
    for (int r = 0; r < 8; ++r) {
      int m  = mt * 16 + half * 8 + r;     // C/D layout: VGPR r, halves +8
      int tl = subt * 64 + m;
      float y = acc[mt][r] + h0v * bf2f(sU.x[ci][bn][tl + 511]);
      acc[mt][r] = gelu_tanh(y);
    }
  }
  __syncthreads();                          // x windows no longer needed
#pragma unroll
  for (int mt = 0; mt < 4; ++mt) {
#pragma unroll
    for (int r = 0; r < 8; ++r) {
      int m  = mt * 16 + half * 8 + r;
      int tl = subt * 64 + m;
      sU.o[(tl * NB + bn) * CHG + ci] = acc[mt][r];
    }
  }
  __syncthreads();
  // Coalesced 32B-run stores into (B, L, C) layout
  for (int rep = 0; rep < 4; ++rep) {
    int row = rep * 256 + tid;              // row = (tl, b)
    int tl = row >> 2, b = row & 3;
    const float4* src = reinterpret_cast<const float4*>(&sU.o[row * CHG]);
    float4* dst = reinterpret_cast<float4*>(
        &out[((size_t)b * LSEQ + (t0 + tl)) * NCH + c0]);
    dst[0] = src[0];
    dst[1] = src[1];
  }
}

// ---------------------------------------------------------------------------
extern "C" void kernel_launch(void* const* d_in, const int* in_sizes, int n_in,
                              void* d_out, int out_size, void* d_ws, size_t ws_size,
                              hipStream_t stream) {
  (void)in_sizes; (void)n_in; (void)out_size; (void)ws_size;
  const float* x  = (const float*)d_in[0];
  const float* A  = (const float*)d_in[1];
  const float* Bc = (const float*)d_in[2];
  const float* h0 = (const float*)d_in[3];
  float* out = (float*)d_out;

  unsigned short* KR = (unsigned short*)d_ws;            // 512*640 bf16
  unsigned short* xT = KR + (size_t)NCH * KRLEN;         // 4*512*8192 bf16

  rtf_build_kernel<<<dim3(NCH / 256), 256, 0, stream>>>(A, Bc, KR);

  dim3 g2(LSEQ / 64, NCH / 64, NB);
  rtf_xpose_kernel<<<g2, 256, 0, stream>>>(x, xT);

  dim3 g3(LSEQ / TSUP, NCH / CHG);
  rtf_conv_kernel<<<g3, 256, 0, stream>>>(xT, KR, h0, out);
}
